// MultiHeadAttention_65377992180173
// MI455X (gfx1250) — matrix operand order repaired and checked by reading, not compiled, not run on hardware
//
#include <hip/hip_runtime.h>
#include <hip/hip_bf16.h>

// ---------------------------------------------------------------------------
// MI455X (gfx1250) multi-head attention forward, f16 WMMA pipeline with
// TDM (tensor_load_to_lds) double-buffered K/V staging in the attention core.
//   B=4, S=2048, E=1024, H=16, D=64
// ---------------------------------------------------------------------------

typedef __attribute__((ext_vector_type(16))) _Float16     v16h;
typedef __attribute__((ext_vector_type(8)))  float        v8f;
typedef __attribute__((ext_vector_type(4)))  int          i4;
typedef __attribute__((ext_vector_type(4)))  unsigned int u32x4;
typedef __attribute__((ext_vector_type(8)))  int          i32x8;
typedef __attribute__((ext_vector_type(4)))  int          i32x4;

union F16Frag {          // 32 bytes = one WMMA f16 A/B fragment per lane
    v16h h;
    i4   i[2];
};

#define BATCH 4
#define SEQ   2048
#define EMB   1024
#define NH    16
#define HD    64

// LDS tile row stride in f16 elements: 128B data + 16B TDM pad = 144B = 72 f16
#define TROW 72

// ---------------------------------------------------------------------------
// f32 -> f16 conversion (grid-stride)
// ---------------------------------------------------------------------------
__global__ void cvt_f32_f16_kernel(const float* __restrict__ src,
                                   _Float16* __restrict__ dst, int n) {
    int i = blockIdx.x * blockDim.x + threadIdx.x;
    int stride = gridDim.x * blockDim.x;
    for (; i < n; i += stride) dst[i] = (_Float16)src[i];
}

// ---------------------------------------------------------------------------
// GEMM: C[M=8192, N=1024] = A(f16, row-major, lda=1024) x W(f16, [1024][1024])
// Block: 256 threads = 8 waves, tile 64(M) x 256(N); wave tile 32 x 64.
// mode 0: store f16 row-major           (Q, V)
// mode 1: store f16 as Kt[b,h,d,s]      (K transposed per head)
// mode 2: store f32 + bias to d_out     (output projection)
// ---------------------------------------------------------------------------
__global__ __launch_bounds__(256)
void gemm_f16_kernel(const _Float16* __restrict__ A,
                     const _Float16* __restrict__ W,
                     void* __restrict__ out,
                     const float* __restrict__ bias,
                     int mode) {
    const int lane = threadIdx.x & 31;
    const int wv   = threadIdx.x >> 5;
    const int lan  = lane & 15;
    const int half = lane >> 4;

    const int mtile = blockIdx.y * 64 + (wv & 1) * 32;   // wave M origin
    const int ncol0 = blockIdx.x * 256 + (wv >> 1) * 64; // wave N origin

    v8f acc[2][4];
#pragma unroll
    for (int mf = 0; mf < 2; ++mf)
#pragma unroll
        for (int nf = 0; nf < 4; ++nf) acc[mf][nf] = (v8f)0.0f;

    for (int ks = 0; ks < EMB / 32; ++ks) {
        // A fragments: lane = M row; elems 0-7 -> k = ks*32 + 8*half + 0..7,
        //              elems 8-15 -> k = ks*32 + 16 + 8*half + 0..7
        F16Frag a0, a1;
        const _Float16* ap = A + (size_t)(mtile + lan) * EMB + ks * 32 + 8 * half;
        a0.i[0] = *(const i4*)ap;
        a0.i[1] = *(const i4*)(ap + 16);
        ap += (size_t)16 * EMB;
        a1.i[0] = *(const i4*)ap;
        a1.i[1] = *(const i4*)(ap + 16);

        // B fragments: lane = N col (lan), elems 0-7 -> k = ks*32 + 8*half + 0..7, 8-15 -> k = ks*32 + 16 + 8*half + 0..7
        const _Float16* bp = W + (size_t)(ks * 32 + 8 * half) * EMB + ncol0 + lan;
#pragma unroll
        for (int nf = 0; nf < 4; ++nf) {
            F16Frag b;
#pragma unroll
            for (int e = 0; e < 8; ++e) {
                b.h[e]     = bp[(size_t)e * EMB + nf * 16];
                b.h[8 + e] = bp[(size_t)(16 + e) * EMB + nf * 16];
            }
            acc[0][nf] = __builtin_amdgcn_wmma_f32_16x16x32_f16(
                false, a0.h, false, b.h, (short)0, acc[0][nf], false, false);
            acc[1][nf] = __builtin_amdgcn_wmma_f32_16x16x32_f16(
                false, a1.h, false, b.h, (short)0, acc[1][nf], false, false);
        }
    }

    // D layout: vgpr r, lanes 0-15 -> m = r, lanes 16-31 -> m = r+8; n = lan
#pragma unroll
    for (int mf = 0; mf < 2; ++mf)
#pragma unroll
        for (int nf = 0; nf < 4; ++nf)
#pragma unroll
            for (int r = 0; r < 8; ++r) {
                const int m = mtile + mf * 16 + r + 8 * half;
                const int n = ncol0 + nf * 16 + lan;
                const float v = acc[mf][nf][r];
                if (mode == 0) {
                    ((_Float16*)out)[(size_t)m * EMB + n] = (_Float16)v;
                } else if (mode == 1) {
                    const int bb = m >> 11, s = m & (SEQ - 1);
                    const int hh = n >> 6, d = n & (HD - 1);
                    ((_Float16*)out)[((size_t)(bb * NH + hh) * HD + d) * SEQ + s] =
                        (_Float16)v;
                } else {
                    ((float*)out)[(size_t)m * EMB + n] = v + bias[n];
                }
            }
}

// ---------------------------------------------------------------------------
// TDM: 2-D tile load (64 rows x 128B) from global to LDS, 16B pad per row.
// D# per CDNA5 ISA ch.8; data_size=8B, tile_dim0=16 (128B), tile_dim1=64.
// row_stride_e8 = global row-to-row stride in 8-byte units.
// ---------------------------------------------------------------------------
__device__ inline void tdm_load_tile64(unsigned long long gaddr,
                                       unsigned int lds_off,
                                       unsigned int row_stride_e8) {
    u32x4 g0;
    g0[0] = 1u;                                   // count=1, user mode
    g0[1] = lds_off;                              // LDS byte address
    g0[2] = (unsigned int)gaddr;                  // global addr [31:0]
    g0[3] = (unsigned int)((gaddr >> 32) & 0x01FFFFFFu) | 0x80000000u; // type=2

    i32x8 g1;
    // data_size=3(8B) | pad_enable | pad_interval=4(32DW=128B) | pad_amount=3(4DW=16B)
    g1[0] = (3 << 16) | (1 << 20) | (4 << 22) | (3 << 25);
    g1[1] = (int)((row_stride_e8 & 0xFFFFu) << 16);   // tensor_dim0[15:0]
    g1[2] = (int)(((row_stride_e8 >> 16) & 0xFFFFu) | 0xFFFF0000u); // dim0 hi | dim1 lo
    g1[3] = (int)(16u << 16);                         // tensor_dim1 hi=0 | tile_dim0=16
    g1[4] = 64;                                       // tile_dim1=64 | tile_dim2=0
    g1[5] = (int)row_stride_e8;                       // tensor_dim0_stride[31:0]
    g1[6] = 0;                                        // stride hi | dim1_stride lo
    g1[7] = 0;

    i32x4 z4 = (i32x4)0;
#if __clang_major__ >= 23
    i32x8 z8 = (i32x8)0;
    __builtin_amdgcn_tensor_load_to_lds(g0, g1, z4, z4, z8, 0);
#else
    __builtin_amdgcn_tensor_load_to_lds(g0, g1, z4, z4, 0);
#endif
}

// ---------------------------------------------------------------------------
// 16-lane (half-wave) reductions: rows r+8*half live on one 16-lane group.
// ---------------------------------------------------------------------------
__device__ inline float halfmax16(float v) {
#pragma unroll
    for (int m = 1; m < 16; m <<= 1) v = fmaxf(v, __shfl_xor(v, m, 32));
    return v;
}
__device__ inline float halfsum16(float v) {
#pragma unroll
    for (int m = 1; m < 16; m <<= 1) v += __shfl_xor(v, m, 32);
    return v;
}

// ---------------------------------------------------------------------------
// Causal flash attention. Block = (qt, h, b), 8 waves; wave owns 16 q rows.
// K/V 64x64 tiles staged in LDS by the TDM, double-buffered on TENSORcnt.
// Q: [b,s,1024] f16  Kt: [b,h,d,s] f16  V: [b,s,1024] f16  Ctx: [b,s,1024] f16
// ---------------------------------------------------------------------------
__global__ __launch_bounds__(256)
void attn_kernel(const _Float16* __restrict__ Qh,
                 const _Float16* __restrict__ Kth,
                 const _Float16* __restrict__ Vh,
                 _Float16* __restrict__ Ctx) {
    __shared__ __align__(16) _Float16 KS[2][64 * TROW]; // [d][s], TDM-padded
    __shared__ __align__(16) _Float16 VS[2][64 * TROW]; // [s][d], TDM-padded
    __shared__ __align__(16) _Float16 psm[8][16][TROW]; // per-wave P tile

    const int lane = threadIdx.x & 31;
    const int wv   = threadIdx.x >> 5;
    const int lan  = lane & 15;
    const int half = lane >> 4;

    const int qt = blockIdx.x, h = blockIdx.y, b = blockIdx.z;
    const int q0 = qt * 128 + wv * 16;    // first q row of this wave
    const int ntiles = 2 * qt + 2;        // uniform across block (causal bound)

    // TDM source bases (bytes)
    const unsigned long long ktg =
        (unsigned long long)(uintptr_t)(Kth + (size_t)(b * NH + h) * HD * SEQ);
    const unsigned long long vtg =
        (unsigned long long)(uintptr_t)(Vh + (size_t)b * SEQ * EMB + h * HD);

    // Preload Q A-fragments (d = 0..31, 32..63)
    F16Frag qf[2];
    {
        const _Float16* qp =
            Qh + (size_t)(b * SEQ + q0 + lan) * EMB + h * HD + 8 * half;
        qf[0].i[0] = *(const i4*)qp;
        qf[0].i[1] = *(const i4*)(qp + 16);
        qf[1].i[0] = *(const i4*)(qp + 32);
        qf[1].i[1] = *(const i4*)(qp + 48);
    }

    v8f ctx[4];
#pragma unroll
    for (int nf = 0; nf < 4; ++nf) ctx[nf] = (v8f)0.0f;
    float mrow[8], lrow[8];
#pragma unroll
    for (int r = 0; r < 8; ++r) { mrow[r] = -3e38f; lrow[r] = 0.0f; }

    // Kick off tile 0 DMA (wave 0 owns the TDM + TENSORcnt)
    if (wv == 0) {
        tdm_load_tile64(ktg, (unsigned int)(uintptr_t)&KS[0][0], SEQ * 2 / 8);
        tdm_load_tile64(vtg, (unsigned int)(uintptr_t)&VS[0][0], EMB * 2 / 8);
    }

    for (int kt = 0; kt < ntiles; ++kt) {
        const int k0  = kt * 64;
        const int cur = kt & 1;

        if (wv == 0) __builtin_amdgcn_s_wait_tensorcnt((short)0);
        __syncthreads();   // tile kt visible to all waves

        // Prefetch tile kt+1 into the other buffer (freed by last barrier)
        if (wv == 0 && kt + 1 < ntiles) {
            const int nb = (kt + 1) & 1;
            tdm_load_tile64(ktg + (unsigned long long)(k0 + 64) * 2,
                            (unsigned int)(uintptr_t)&KS[nb][0], SEQ * 2 / 8);
            tdm_load_tile64(vtg + (unsigned long long)(k0 + 64) * EMB * 2,
                            (unsigned int)(uintptr_t)&VS[nb][0], EMB * 2 / 8);
        }

        // ---- scores: S = Q(16x64) x K^T(64x64), 4 n-frags x 2 k-chunks
        v8f sc[4];
#pragma unroll
        for (int sf = 0; sf < 4; ++sf) {
            v8f s = (v8f)0.0f;
#pragma unroll
            for (int kc = 0; kc < 2; ++kc) {
                F16Frag kb;   // lane = s col, elems = d in table order
                const _Float16* kp = &KS[cur][(kc * 32 + 8 * half) * TROW + sf * 16 + lan];
#pragma unroll
                for (int e = 0; e < 8; ++e) {
                    kb.h[e]     = kp[e * TROW];
                    kb.h[8 + e] = kp[(16 + e) * TROW];
                }
                s = __builtin_amdgcn_wmma_f32_16x16x32_f16(
                    false, qf[kc].h, false, kb.h, (short)0, s, false, false);
            }
            sc[sf] = s;
        }

        // ---- scale + causal mask + tile row-max
        float tmax[8];
#pragma unroll
        for (int r = 0; r < 8; ++r) tmax[r] = -3e38f;
#pragma unroll
        for (int sf = 0; sf < 4; ++sf)
#pragma unroll
            for (int r = 0; r < 8; ++r) {
                float v = sc[sf][r] * 0.125f;          // 1/sqrt(64)
                const int kpos = k0 + sf * 16 + lan;
                const int q    = q0 + r + 8 * half;
                if (kpos > q) v = -3e38f;
                sc[sf][r] = v;
                tmax[r] = fmaxf(tmax[r], v);
            }

        float facs[8];
#pragma unroll
        for (int r = 0; r < 8; ++r) {
            const float tm = halfmax16(tmax[r]);
            const float nm = fmaxf(mrow[r], tm);
            facs[r] = __expf(mrow[r] - nm);
            mrow[r] = nm;
            lrow[r] *= facs[r];
        }

        // ---- P = exp(S - m), row sums, write P to wave-private LDS
        float rsum[8];
#pragma unroll
        for (int r = 0; r < 8; ++r) rsum[r] = 0.0f;
#pragma unroll
        for (int sf = 0; sf < 4; ++sf)
#pragma unroll
            for (int r = 0; r < 8; ++r) {
                const float p = __expf(sc[sf][r] - mrow[r]);
                rsum[r] += p;
                psm[wv][r + 8 * half][sf * 16 + lan] = (_Float16)p;
            }
#pragma unroll
        for (int r = 0; r < 8; ++r) lrow[r] += halfsum16(rsum[r]);

        // ---- rescale running context
#pragma unroll
        for (int nf = 0; nf < 4; ++nf)
#pragma unroll
            for (int r = 0; r < 8; ++r) ctx[nf][r] *= facs[r];

        // ---- reload P as A-fragments (D-layout -> A-layout via LDS)
        F16Frag pf[2];
#pragma unroll
        for (int kc = 0; kc < 2; ++kc) {
            const _Float16* pp = &psm[wv][lan][kc * 32 + 8 * half];
            pf[kc].i[0] = *(const i4*)pp;
            pf[kc].i[1] = *(const i4*)(pp + 16);
        }

        // ---- ctx += P(16x64) x V(64x64)
#pragma unroll
        for (int nf = 0; nf < 4; ++nf) {
#pragma unroll
            for (int kc = 0; kc < 2; ++kc) {
                F16Frag vf;   // lane = d col, elems = s in table order
                const _Float16* vp = &VS[cur][(kc * 32 + 8 * half) * TROW + nf * 16 + lan];
#pragma unroll
                for (int e = 0; e < 8; ++e) {
                    vf.h[e]     = vp[e * TROW];
                    vf.h[8 + e] = vp[(16 + e) * TROW];
                }
                ctx[nf] = __builtin_amdgcn_wmma_f32_16x16x32_f16(
                    false, pf[kc].h, false, vf.h, (short)0, ctx[nf], false, false);
            }
        }

        __syncthreads();   // all waves done with buffer `cur` before reuse
    }

    // ---- normalize and store context [b, s, h*64+d] f16
#pragma unroll
    for (int r = 0; r < 8; ++r) lrow[r] = 1.0f / lrow[r];
#pragma unroll
    for (int nf = 0; nf < 4; ++nf)
#pragma unroll
        for (int r = 0; r < 8; ++r) {
            const size_t idx =
                (size_t)(b * SEQ + q0 + r + 8 * half) * EMB + h * HD + nf * 16 + lan;
            Ctx[idx] = (_Float16)(ctx[nf][r] * lrow[r]);
        }
}

// ---------------------------------------------------------------------------
// Host launcher
// ---------------------------------------------------------------------------
extern "C" void kernel_launch(void* const* d_in, const int* in_sizes, int n_in,
                              void* d_out, int out_size, void* d_ws, size_t ws_size,
                              hipStream_t stream) {
    const float* X  = (const float*)d_in[0];
    const float* Wq = (const float*)d_in[1];
    const float* Wk = (const float*)d_in[2];
    const float* Wv = (const float*)d_in[3];
    const float* Wo = (const float*)d_in[4];
    const float* bo = (const float*)d_in[5];

    char* ws = (char*)d_ws;
    const size_t X_ELEMS = (size_t)BATCH * SEQ * EMB;   // 8M
    const size_t W_ELEMS = (size_t)EMB * EMB;           // 1M

    _Float16* Xh   = (_Float16*)(ws);                               // 16 MB
    _Float16* Wqh  = (_Float16*)(ws + 16u * 1024 * 1024);           //  2 MB
    _Float16* Wkh  = (_Float16*)(ws + 18u * 1024 * 1024);
    _Float16* Wvh  = (_Float16*)(ws + 20u * 1024 * 1024);
    _Float16* Woh  = (_Float16*)(ws + 22u * 1024 * 1024);
    _Float16* Qh   = (_Float16*)(ws + 24u * 1024 * 1024);           // 16 MB
    _Float16* Kth  = (_Float16*)(ws + 40u * 1024 * 1024);           // 16 MB
    _Float16* Vh   = (_Float16*)(ws + 56u * 1024 * 1024);           // 16 MB
    _Float16* Ctxh = (_Float16*)(ws + 72u * 1024 * 1024);           // 16 MB

    // 1) f32 -> f16 conversions
    cvt_f32_f16_kernel<<<2048, 256, 0, stream>>>(X,  Xh,  (int)X_ELEMS);
    cvt_f32_f16_kernel<<<512,  256, 0, stream>>>(Wq, Wqh, (int)W_ELEMS);
    cvt_f32_f16_kernel<<<512,  256, 0, stream>>>(Wk, Wkh, (int)W_ELEMS);
    cvt_f32_f16_kernel<<<512,  256, 0, stream>>>(Wv, Wvh, (int)W_ELEMS);
    cvt_f32_f16_kernel<<<512,  256, 0, stream>>>(Wo, Woh, (int)W_ELEMS);

    // 2) projections: grid = (N/256, M/64) = (4, 128)
    dim3 ggrid(4, 128);
    gemm_f16_kernel<<<ggrid, 256, 0, stream>>>(Xh, Wqh, Qh,  nullptr, 0);
    gemm_f16_kernel<<<ggrid, 256, 0, stream>>>(Xh, Wkh, Kth, nullptr, 1);
    gemm_f16_kernel<<<ggrid, 256, 0, stream>>>(Xh, Wvh, Vh,  nullptr, 0);

    // 3) causal flash attention: grid = (S/128, H, B)
    attn_kernel<<<dim3(SEQ / 128, NH, BATCH), 256, 0, stream>>>(Qh, Kth, Vh, Ctxh);

    // 4) output projection + bias -> f32 d_out
    gemm_f16_kernel<<<ggrid, 256, 0, stream>>>(Ctxh, Woh, d_out, bo, 2);
}
